// MultiQueryAttention_84353157693678
// MI455X (gfx1250) — matrix operand order repaired and checked by reading, not compiled, not run on hardware
//
#include <hip/hip_runtime.h>

// MQA forward for MI455X (gfx1250, wave32).
// bf16 WMMA (v_wmma_f32_16x16x32_bf16) for all matmuls, f32 accumulate,
// flash-attention online softmax, final output in f32.
// gfx1250 data movement: global_prefetch_b8 for next-tile prefetch and
// global_load_async_to_lds_b128 (ASYNCcnt) for bf16 A-tile staging.
// Fragments are loaded in batches ahead of the WMMA groups so the scheduler
// can clause the loads and overlap them with matrix math.

#define D_MODEL 1024
#define SEQ_N   2048
#define BATCH   4
#define NHEADS  16
#define DK      64

#if defined(__has_builtin)
#if __has_builtin(__builtin_amdgcn_global_load_async_to_lds_b128)
#define HAVE_ASYNC_LDS 1
#endif
#endif
#ifndef HAVE_ASYNC_LDS
#define HAVE_ASYNC_LDS 0
#endif

typedef __attribute__((ext_vector_type(16))) __bf16 v16bf;
typedef __attribute__((ext_vector_type(8)))  float  v8f;

#if HAVE_ASYNC_LDS
// Exact pointee type expected by the async-to-LDS builtin (from its
// diagnostic): GCC-vector int4, global (AS1) source, LDS (AS3) destination.
typedef int gcc_v4i __attribute__((__vector_size__(16)));
typedef __attribute__((address_space(1))) gcc_v4i* g_v4i_ptr;
typedef __attribute__((address_space(3))) gcc_v4i* l_v4i_ptr;
#endif

union BF16Frag {
    v16bf v;
    unsigned int u[8];
    unsigned short h[16];
};

__device__ __forceinline__ unsigned short f2bf(float f) {
    union { float f; unsigned int u; } x; x.f = f;
    unsigned int r = x.u + 0x7FFFu + ((x.u >> 16) & 1u);  // RNE
    return (unsigned short)(r >> 16);
}

__device__ __forceinline__ v8f wmma_bf16(v16bf a, v16bf b, v8f c) {
    // (neg_a, A, neg_b, B, c_mod, C, reuse_a, reuse_b)
    return __builtin_amdgcn_wmma_f32_16x16x32_bf16(false, a, false, b, (short)0, c, false, false);
}

__device__ __forceinline__ float hmax16(float x) {
    #pragma unroll
    for (int m = 1; m < 16; m <<= 1) x = fmaxf(x, __shfl_xor(x, m, 32));
    return x;
}
__device__ __forceinline__ float hsum16(float x) {
    #pragma unroll
    for (int m = 1; m < 16; m <<= 1) x += __shfl_xor(x, m, 32);
    return x;
}

// ---------------------------------------------------------------------------
// f32 -> bf16 weight conversion
// ---------------------------------------------------------------------------
__global__ void cvt_f32_bf16(const float* __restrict__ in,
                             unsigned short* __restrict__ out, int n) {
    int i = blockIdx.x * blockDim.x + threadIdx.x;
    if (i < n) out[i] = f2bf(in[i]);
}

// ---------------------------------------------------------------------------
// Tiled WMMA GEMM: C[M x Nw] = A[M x K] @ W[K x Nw]
// macro tile 128(M) x 64(N), 128 threads = 4 waves, each wave owns 32 rows:
// per K-step: 2 A-fragments + 4 shared B-fragments -> 8 WMMAs.
// MODE 0: A=f32, out=bf16 row-major stride Nw              (K/V projections)
// MODE 1: A=f32, out=bf16 scattered head-major q[b,h,n,d]  (Q projection)
// MODE 2: A=bf16 (async-to-LDS staged), out=f32            (output projection)
// ---------------------------------------------------------------------------
template <int MODE>
__global__ __launch_bounds__(128) void gemm_bf16_wmma(
    const void* __restrict__ Av, const unsigned short* __restrict__ W,
    void* __restrict__ Outv, int K, int Nw) {
    __shared__ unsigned short lA[128 * 32];   // [row][k] bf16, 8KB
    __shared__ unsigned short lWt[64 * 32];   // [col][k] bf16 (transposed), 4KB

    const int tid   = threadIdx.x;            // 0..127
    const int lane  = tid & 31;
    const int wave  = tid >> 5;               // 0..3, owns rows wave*32..+31
    const int khalf = lane >> 4;
    const int l16   = lane & 15;
    const int m0    = blockIdx.x * 128;
    const int n0    = blockIdx.y * 64;

    const v8f vzero = {0.f, 0.f, 0.f, 0.f, 0.f, 0.f, 0.f, 0.f};
    v8f acc[2][4];
    #pragma unroll
    for (int hh = 0; hh < 2; ++hh)
        #pragma unroll
        for (int nt = 0; nt < 4; ++nt) acc[hh][nt] = vzero;

    for (int k0 = 0; k0 < K; k0 += 32) {
        // ---- prefetch next K-tile into cache while we stage this one ----
        if (k0 + 32 < K) {
            if (MODE == 2) {
                const unsigned short* A = (const unsigned short*)Av;
                __builtin_prefetch(&A[(size_t)(m0 + tid) * K + k0 + 32], 0, 1);
            } else {
                const float* A = (const float*)Av;
                __builtin_prefetch(&A[(size_t)(m0 + tid) * K + k0 + 32], 0, 1);
            }
            if (tid < 32)
                __builtin_prefetch(&W[(size_t)(k0 + 32 + tid) * Nw + n0], 0, 1);
        }

        // ---- stage A tile ----
        if (MODE == 2) {
            const unsigned short* A = (const unsigned short*)Av;
#if HAVE_ASYNC_LDS
            // bf16 rows are 64B contiguous both in memory and LDS:
            // DMA memory->LDS, 16B per lane, 4 segments per row.
            for (int i = tid; i < 512; i += 128) {
                int r = i >> 2, seg = (i & 3) * 8;   // row, ushort offset
                __builtin_amdgcn_global_load_async_to_lds_b128(
                    (g_v4i_ptr)&A[(size_t)(m0 + r) * K + k0 + seg],
                    (l_v4i_ptr)&lA[r * 32 + seg],
                    0, 0);
            }
            asm volatile("s_wait_asynccnt 0x0" ::: "memory");
#else
            for (int i = tid; i < 128 * 32; i += 128) {
                int r = i >> 5, kk = i & 31;
                lA[i] = A[(size_t)(m0 + r) * K + k0 + kk];
            }
#endif
        } else {
            const float* A = (const float*)Av;
            for (int i = tid; i < 128 * 32; i += 128) {
                int r = i >> 5, kk = i & 31;
                lA[i] = f2bf(A[(size_t)(m0 + r) * K + k0 + kk]);
            }
        }
        // ---- stage W tile transposed: lWt[col][k] ----
        for (int i = tid; i < 64 * 32; i += 128) {
            int kk = i >> 6, c = i & 63;
            lWt[c * 32 + kk] = W[(size_t)(k0 + kk) * Nw + n0 + c];
        }
        __syncthreads();

        // A fragments (16-bit A 16x32 layout): lane holds row l16,
        // K = (v>=4?16:0) + khalf*8 + (v%4)*2
        BF16Frag a0, a1;
        #pragma unroll
        for (int v = 0; v < 8; ++v) {
            int kk = ((v & 4) ? 16 : 0) + khalf * 8 + (v & 3) * 2;
            a0.u[v] = *(const unsigned int*)&lA[(wave * 32 + l16) * 32 + kk];
            a1.u[v] = *(const unsigned int*)&lA[(wave * 32 + 16 + l16) * 32 + kk];
        }
        // All 4 B fragments first (lane holds col l16, K = (v>=4?16:0) + khalf*8 + (v%4)*2),
        // then 8 WMMAs: lets the scheduler clause the ds loads.
        BF16Frag bfr[4];
        #pragma unroll
        for (int nt = 0; nt < 4; ++nt)
            #pragma unroll
            for (int v = 0; v < 8; ++v)
                bfr[nt].u[v] = *(const unsigned int*)&lWt[(nt * 16 + l16) * 32 + ((v & 4) ? 16 : 0) + khalf * 8 + (v & 3) * 2];
        #pragma unroll
        for (int nt = 0; nt < 4; ++nt) {
            acc[0][nt] = wmma_bf16(a0.v, bfr[nt].v, acc[0][nt]);
            acc[1][nt] = wmma_bf16(a1.v, bfr[nt].v, acc[1][nt]);
        }
        __syncthreads();
    }

    // C layout: VGPR r -> M = r + 8*(lane/16), N = lane%16
    #pragma unroll
    for (int hh = 0; hh < 2; ++hh) {
        #pragma unroll
        for (int nt = 0; nt < 4; ++nt) {
            #pragma unroll
            for (int r = 0; r < 8; ++r) {
                int mg = m0 + wave * 32 + hh * 16 + r + 8 * khalf;
                int cg = n0 + nt * 16 + l16;
                float val = acc[hh][nt][r];
                if (MODE == 0) {
                    ((unsigned short*)Outv)[(size_t)mg * Nw + cg] = f2bf(val);
                } else if (MODE == 1) {
                    int b = mg >> 11, n = mg & (SEQ_N - 1);
                    int h = cg >> 6, d = cg & (DK - 1);
                    ((unsigned short*)Outv)[(((size_t)b * NHEADS + h) * SEQ_N + n) * DK + d] = f2bf(val);
                } else {
                    ((float*)Outv)[(size_t)mg * D_MODEL + cg] = val;
                }
            }
        }
    }
}

// ---------------------------------------------------------------------------
// Causal flash attention, multi-query (shared K/V head).
// 1 wave = one 16-query tile of one (b,h); 4 waves per block.
// Per 32-key block: 4 WMMAs for scores, online softmax, 4 WMMAs for PV.
// V fragments are fetched right after the score WMMAs so their global
// latency hides under the softmax VALU work and the LDS P-transpose.
// ---------------------------------------------------------------------------
__global__ __launch_bounds__(128) void mqa_attn(
    const unsigned short* __restrict__ qb,   // [b,h,n,64] bf16
    const unsigned short* __restrict__ kb,   // [b,n,64]   bf16
    const unsigned short* __restrict__ vb,   // [b,n,64]   bf16
    unsigned short* __restrict__ attnb) {    // [b,n,1024] bf16
    __shared__ unsigned short lP[4][16 * 32];  // per-wave P tile (16q x 32k)

    const int tid   = threadIdx.x;
    const int lane  = tid & 31;
    const int wslot = tid >> 5;
    const int khalf = lane >> 4;
    const int l16   = lane & 15;

    const int w    = blockIdx.x * 4 + wslot;      // 0..8191
    const int b    = w >> 11;
    const int h    = (w >> 7) & (NHEADS - 1);
    const int tile = w & 127;
    const int n0   = tile * 16;

    const size_t qbase  = (((size_t)b * NHEADS + h) * SEQ_N + n0) * DK;
    const size_t kvbase = (size_t)b * SEQ_N * DK;

    // Q fragments for K-dims 0..31 and 32..63 (kept in VGPRs whole kernel)
    BF16Frag qf0, qf1;
    #pragma unroll
    for (int v = 0; v < 8; ++v) {
        int kk = khalf * 16 + v * 2;
        qf0.u[v] = *(const unsigned int*)&qb[qbase + (size_t)l16 * DK + kk];
        qf1.u[v] = *(const unsigned int*)&qb[qbase + (size_t)l16 * DK + kk + 32];
    }

    const v8f vzero = {0.f, 0.f, 0.f, 0.f, 0.f, 0.f, 0.f, 0.f};
    float m_i[8], l_i[8];
    v8f acc[4];
    #pragma unroll
    for (int r = 0; r < 8; ++r) { m_i[r] = -1e30f; l_i[r] = 0.f; }
    #pragma unroll
    for (int nt = 0; nt < 4; ++nt) acc[nt] = vzero;

    const int nkb = ((n0 + 16) + 31) >> 5;  // causal: keys 0..n0+15
    for (int jb = 0; jb < nkb; ++jb) {
        const int kb0 = jb * 32;

        // prefetch next key block's K and V rows (one touch per row)
        if (jb + 1 < nkb) {
            __builtin_prefetch(&kb[kvbase + (size_t)(kb0 + 32 + lane) * DK], 0, 1);
            __builtin_prefetch(&vb[kvbase + (size_t)(kb0 + 32 + lane) * DK], 0, 1);
        }

        // ---- all 4 K-fragments up front, then 4 score WMMAs ----
        // B = K^T: col=key(lane%16), K=dim=khalf*16+2v
        BF16Frag kf00, kf01, kf10, kf11;
        #pragma unroll
        for (int v = 0; v < 8; ++v) {
            int kd = khalf * 16 + v * 2;
            kf00.u[v] = *(const unsigned int*)&kb[kvbase + (size_t)(kb0 + l16) * DK + kd];
            kf01.u[v] = *(const unsigned int*)&kb[kvbase + (size_t)(kb0 + l16) * DK + kd + 32];
            kf10.u[v] = *(const unsigned int*)&kb[kvbase + (size_t)(kb0 + 16 + l16) * DK + kd];
            kf11.u[v] = *(const unsigned int*)&kb[kvbase + (size_t)(kb0 + 16 + l16) * DK + kd + 32];
        }
        v8f s0 = vzero, s1 = vzero;   // keys kb0..+15 and kb0+16..+31
        s0 = wmma_bf16(qf0.v, kf00.v, s0);
        s0 = wmma_bf16(qf1.v, kf01.v, s0);
        s1 = wmma_bf16(qf0.v, kf10.v, s1);
        s1 = wmma_bf16(qf1.v, kf11.v, s1);

        // ---- V fragments early: latency hides under softmax + LDS ----
        // B = V block [32 keys x 64 dims], col=dim, K=key
        BF16Frag vf[4];
        #pragma unroll
        for (int nt = 0; nt < 4; ++nt) {
            #pragma unroll
            for (int v = 0; v < 8; ++v) {
                int keyl = khalf * 16 + 2 * v;
                vf[nt].h[2 * v]     = vb[kvbase + (size_t)(kb0 + keyl) * DK + nt * 16 + l16];
                vf[nt].h[2 * v + 1] = vb[kvbase + (size_t)(kb0 + keyl + 1) * DK + nt * 16 + l16];
            }
        }

        // scale + causal mask + online softmax (rows live in 16-lane halves)
        float p0[8], p1[8], alpha[8];
        #pragma unroll
        for (int r = 0; r < 8; ++r) {
            int mg   = n0 + r + 8 * khalf;
            int key0 = kb0 + l16;
            int key1 = kb0 + 16 + l16;
            float v0 = (key0 <= mg) ? s0[r] * 0.125f : -1e30f;
            float v1 = (key1 <= mg) ? s1[r] * 0.125f : -1e30f;
            float bm = hmax16(fmaxf(v0, v1));
            float nm = fmaxf(m_i[r], bm);
            alpha[r] = __expf(m_i[r] - nm);
            m_i[r]   = nm;
            p0[r] = __expf(v0 - nm);
            p1[r] = __expf(v1 - nm);
            l_i[r] = l_i[r] * alpha[r] + hsum16(p0[r] + p1[r]);
        }
        #pragma unroll
        for (int nt = 0; nt < 4; ++nt)
            #pragma unroll
            for (int r = 0; r < 8; ++r)
                acc[nt][r] *= alpha[r];

        // P (C-layout) -> LDS -> A-fragment layout
        #pragma unroll
        for (int r = 0; r < 8; ++r) {
            int row = r + 8 * khalf;
            lP[wslot][row * 32 + l16]      = f2bf(p0[r]);
            lP[wslot][row * 32 + 16 + l16] = f2bf(p1[r]);
        }
        asm volatile("s_wait_dscnt 0x0" ::: "memory");  // in-wave LDS RAW
        BF16Frag pf;
        #pragma unroll
        for (int v = 0; v < 8; ++v) {
            int kk = khalf * 16 + v * 2;
            pf.u[v] = *(const unsigned int*)&lP[wslot][l16 * 32 + kk];
        }

        #pragma unroll
        for (int nt = 0; nt < 4; ++nt)
            acc[nt] = wmma_bf16(pf.v, vf[nt].v, acc[nt]);
    }

    // epilogue: normalize and store bf16 context [b, n, h*64+d]
    #pragma unroll
    for (int nt = 0; nt < 4; ++nt) {
        #pragma unroll
        for (int r = 0; r < 8; ++r) {
            int mg  = n0 + r + 8 * khalf;
            int col = h * DK + nt * 16 + l16;
            attnb[((size_t)b * SEQ_N + mg) * D_MODEL + col] = f2bf(acc[nt][r] / l_i[r]);
        }
    }
}

// ---------------------------------------------------------------------------
extern "C" void kernel_launch(void* const* d_in, const int* in_sizes, int n_in,
                              void* d_out, int out_size, void* d_ws, size_t ws_size,
                              hipStream_t stream) {
    const float* Q   = (const float*)d_in[0];
    const float* Kin = (const float*)d_in[1];
    const float* Vin = (const float*)d_in[2];
    const float* w_q = (const float*)d_in[3];
    const float* w_k = (const float*)d_in[4];
    const float* w_v = (const float*)d_in[5];
    const float* w_o = (const float*)d_in[6];

    // workspace layout (bf16 as ushort), ~40 MB total
    unsigned short* p = (unsigned short*)d_ws;
    unsigned short* wq_bf = p; p += (size_t)D_MODEL * D_MODEL;
    unsigned short* wk_bf = p; p += (size_t)D_MODEL * DK;
    unsigned short* wv_bf = p; p += (size_t)D_MODEL * DK;
    unsigned short* wo_bf = p; p += (size_t)D_MODEL * D_MODEL;
    unsigned short* qb    = p; p += (size_t)BATCH * SEQ_N * D_MODEL;
    unsigned short* kbuf  = p; p += (size_t)BATCH * SEQ_N * DK;
    unsigned short* vbuf  = p; p += (size_t)BATCH * SEQ_N * DK;
    unsigned short* attnb = p; p += (size_t)BATCH * SEQ_N * D_MODEL;

    cvt_f32_bf16<<<(D_MODEL * D_MODEL + 255) / 256, 256, 0, stream>>>(w_q, wq_bf, D_MODEL * D_MODEL);
    cvt_f32_bf16<<<(D_MODEL * DK + 255) / 256, 256, 0, stream>>>(w_k, wk_bf, D_MODEL * DK);
    cvt_f32_bf16<<<(D_MODEL * DK + 255) / 256, 256, 0, stream>>>(w_v, wv_bf, D_MODEL * DK);
    cvt_f32_bf16<<<(D_MODEL * D_MODEL + 255) / 256, 256, 0, stream>>>(w_o, wo_bf, D_MODEL * D_MODEL);

    const int Mrows = BATCH * SEQ_N;              // 8192
    dim3 gq(Mrows / 128, D_MODEL / 64);           // (64,16)
    dim3 gkv(Mrows / 128, 1);                     // (64,1)

    gemm_bf16_wmma<1><<<gq, 128, 0, stream>>>(Q, wq_bf, qb, D_MODEL, D_MODEL);
    gemm_bf16_wmma<0><<<gkv, 128, 0, stream>>>(Kin, wk_bf, kbuf, D_MODEL, DK);
    gemm_bf16_wmma<0><<<gkv, 128, 0, stream>>>(Vin, wv_bf, vbuf, D_MODEL, DK);

    mqa_attn<<<(BATCH * NHEADS * (SEQ_N / 16)) / 4, 128, 0, stream>>>(qb, kbuf, vbuf, attnb);

    gemm_bf16_wmma<2><<<gq, 128, 0, stream>>>(attnb, wo_bf, d_out, D_MODEL, D_MODEL);
}
